// Decoder_Block_Color_81295140978728
// MI455X (gfx1250) — compile-verified, matrix operand order checked by reading, not run on hardware
//
#include <hip/hip_runtime.h>
#include <hip/hip_bf16.h>
#include <math.h>

// ---------------------------------------------------------------------------
// Problem constants (from reference)
// ---------------------------------------------------------------------------
#define Bb   16
#define Nn   1024
#define Ll   313
#define Cc   768
#define Hh   12
#define MLPd 3072
#define DHd  64
#define NLt  1337          // N + L
#define NLP  1344          // NL padded to multiple of 32 (v_t token stride)
#define EPSf 1e-5f

typedef __bf16 bf16_t;
typedef __attribute__((ext_vector_type(16))) __bf16 v16bf;
typedef __attribute__((ext_vector_type(8)))  __bf16 v8bf;
typedef __attribute__((ext_vector_type(8)))  float  v8f;
typedef __attribute__((ext_vector_type(4)))  int    v4i;

#define AS1 __attribute__((address_space(1)))
#define AS3 __attribute__((address_space(3)))

#if defined(__gfx1250__) && \
    __has_builtin(__builtin_amdgcn_global_load_async_to_lds_b128) && \
    __has_builtin(__builtin_amdgcn_s_wait_asynccnt)
#define HAS_ASYNC_LDS 1
#else
#define HAS_ASYNC_LDS 0
#endif

// ---------------------------------------------------------------------------
// WMMA helpers (CDNA5 wave32, V_WMMA_F32_16X16X32_BF16)
// A fragment: lane l holds row m=l%16, chunks at k=(l/16)*8 and k=(l/16)*8+16.
// B fragment: lane l holds col n=l%16, same k pattern (so we feed Bt[n][k]).
// D fragment: lane l holds D[m=r+8*(l/16)][n=l%16] in element r of v8f.
// ---------------------------------------------------------------------------
static __device__ __forceinline__ v8bf ld8(const bf16_t* p) {
  return *reinterpret_cast<const v8bf*>(p);
}
static __device__ __forceinline__ v16bf cat16(v8bf lo, v8bf hi) {
  return __builtin_shufflevector(lo, hi, 0,1,2,3,4,5,6,7,8,9,10,11,12,13,14,15);
}
static __device__ __forceinline__ v8f wmma_bf16(v16bf a, v16bf b, v8f c) {
  return __builtin_amdgcn_wmma_f32_16x16x32_bf16(false, a, false, b, (short)0, c,
                                                 false, false);
}
// One wave computes a 16x64 tile: A fragment reused across 4 column subtiles.
// Arow = A + m*lda + kb ; Bd = Bt + (col_d)*ldb + kb ; kb = (lane/16)*8.
static __device__ __forceinline__ void gemm4(const bf16_t* Arow,
                                             const bf16_t* B0, const bf16_t* B1,
                                             const bf16_t* B2, const bf16_t* B3,
                                             int K, v8f acc[4]) {
  for (int k0 = 0; k0 < K; k0 += 32) {
    v16bf a = cat16(ld8(Arow + k0), ld8(Arow + k0 + 16));
    v16bf b0 = cat16(ld8(B0 + k0), ld8(B0 + k0 + 16));
    acc[0] = wmma_bf16(a, b0, acc[0]);
    v16bf b1 = cat16(ld8(B1 + k0), ld8(B1 + k0 + 16));
    acc[1] = wmma_bf16(a, b1, acc[1]);
    v16bf b2 = cat16(ld8(B2 + k0), ld8(B2 + k0 + 16));
    acc[2] = wmma_bf16(a, b2, acc[2]);
    v16bf b3 = cat16(ld8(B3 + k0), ld8(B3 + k0 + 16));
    acc[3] = wmma_bf16(a, b3, acc[3]);
  }
}

// ---------------------------------------------------------------------------
// LayerNorm 1: x -> xn (f32), xn (bf16), tokens[0:N] (bf16)
// ---------------------------------------------------------------------------
__global__ __launch_bounds__(256)
void ln1_kernel(const float* __restrict__ x, const float* __restrict__ g,
                const float* __restrict__ b, float* __restrict__ xn,
                bf16_t* __restrict__ xnb, bf16_t* __restrict__ tokens) {
  int row = blockIdx.x;
  int tid = threadIdx.x;
  const float* xr = x + (size_t)row * Cc;
  __shared__ float red[256];
  float s = 0.f;
  for (int i = tid; i < Cc; i += 256) s += xr[i];
  red[tid] = s; __syncthreads();
  for (int o = 128; o > 0; o >>= 1) { if (tid < o) red[tid] += red[tid + o]; __syncthreads(); }
  float mu = red[0] / (float)Cc;
  __syncthreads();
  float v = 0.f;
  for (int i = tid; i < Cc; i += 256) { float d = xr[i] - mu; v += d * d; }
  red[tid] = v; __syncthreads();
  for (int o = 128; o > 0; o >>= 1) { if (tid < o) red[tid] += red[tid + o]; __syncthreads(); }
  float rstd = rsqrtf(red[0] / (float)Cc + EPSf);
  int bb = row >> 10, n = row & 1023;
  float*  xo = xn  + (size_t)row * Cc;
  bf16_t* xb = xnb + (size_t)row * Cc;
  bf16_t* tr = tokens + ((size_t)bb * NLt + n) * Cc;
  for (int i = tid; i < Cc; i += 256) {
    float yv = (xr[i] - mu) * rstd * g[i] + b[i];
    xo[i] = yv;
    bf16_t bv = (bf16_t)yv;
    xb[i] = bv;
    tr[i] = bv;
  }
}

// LayerNorm 2: x1 -> h (bf16)
__global__ __launch_bounds__(256)
void ln2_kernel(const float* __restrict__ x1, const float* __restrict__ g,
                const float* __restrict__ b, bf16_t* __restrict__ h) {
  int row = blockIdx.x;
  int tid = threadIdx.x;
  const float* xr = x1 + (size_t)row * Cc;
  __shared__ float red[256];
  float s = 0.f;
  for (int i = tid; i < Cc; i += 256) s += xr[i];
  red[tid] = s; __syncthreads();
  for (int o = 128; o > 0; o >>= 1) { if (tid < o) red[tid] += red[tid + o]; __syncthreads(); }
  float mu = red[0] / (float)Cc;
  __syncthreads();
  float v = 0.f;
  for (int i = tid; i < Cc; i += 256) { float d = xr[i] - mu; v += d * d; }
  red[tid] = v; __syncthreads();
  for (int o = 128; o > 0; o >>= 1) { if (tid < o) red[tid] += red[tid + o]; __syncthreads(); }
  float rstd = rsqrtf(red[0] / (float)Cc + EPSf);
  bf16_t* hr = h + (size_t)row * Cc;
  for (int i = tid; i < Cc; i += 256)
    hr[i] = (bf16_t)((xr[i] - mu) * rstd * g[i] + b[i]);
}

// color_emb -> tokens[N:N+L] (bf16)
__global__ __launch_bounds__(256)
void color_kernel(const float* __restrict__ ce, bf16_t* __restrict__ tokens) {
  int idx = blockIdx.x * 256 + threadIdx.x;
  if (idx >= Bb * Ll * Cc) return;
  int c = idx % Cc;
  int rest = idx / Cc;
  int l = rest % Ll, bb = rest / Ll;
  tokens[((size_t)bb * NLt + Nn + l) * Cc + c] = (bf16_t)ce[idx];
}

// W[K][Nc] f32 -> Wt[Nc][K] bf16
__global__ __launch_bounds__(256)
void convw_kernel(const float* __restrict__ W, bf16_t* __restrict__ Wt,
                  int K, int Nc) {
  int idx = blockIdx.x * 256 + threadIdx.x;
  if (idx >= K * Nc) return;
  int k = idx / Nc, n = idx % Nc;
  Wt[(size_t)n * K + k] = (bf16_t)W[idx];
}

// ---------------------------------------------------------------------------
// GEMM kernels: one wave per 16x64 output tile (A fragment reused 4x).
// ---------------------------------------------------------------------------
// Q = xn @ Wq + bq -> q[b,h,n,dh] bf16
__global__ __launch_bounds__(32)
void gemm_q_kernel(const bf16_t* __restrict__ A, const bf16_t* __restrict__ Wt,
                   const float* __restrict__ bias, bf16_t* __restrict__ q) {
  int lane = threadIdx.x, hv = lane >> 4, idx = lane & 15, kb = hv * 8;
  int ct = blockIdx.x, rt = blockIdx.y;
  v8f z = {0.f,0.f,0.f,0.f,0.f,0.f,0.f,0.f};
  v8f acc[4] = {z, z, z, z};
  const bf16_t* Arow  = A  + (size_t)(rt * 16 + idx) * Cc + kb;
  const bf16_t* Bbase = Wt + (size_t)(ct * 64 + idx) * Cc + kb;
  gemm4(Arow, Bbase, Bbase + 16 * Cc, Bbase + 32 * Cc, Bbase + 48 * Cc, Cc, acc);
#pragma unroll
  for (int d = 0; d < 4; d++) {
    int c = ct * 64 + d * 16 + idx;
    float bv = bias[c];
    int hh = c >> 6, dh = c & 63;
#pragma unroll
    for (int r = 0; r < 8; r++) {
      int R = rt * 16 + r + 8 * hv;
      int bb = R >> 10, n = R & 1023;
      q[(((size_t)(bb * Hh + hh) * Nn + n) * DHd + dh)] = (bf16_t)(acc[d][r] + bv);
    }
  }
}

// K = tokens @ Wk + bk -> k[b,h,t,dh] bf16   (rows = B*NL, divisible by 16)
__global__ __launch_bounds__(32)
void gemm_k_kernel(const bf16_t* __restrict__ A, const bf16_t* __restrict__ Wt,
                   const float* __restrict__ bias, bf16_t* __restrict__ kk) {
  int lane = threadIdx.x, hv = lane >> 4, idx = lane & 15, kb = hv * 8;
  int ct = blockIdx.x, rt = blockIdx.y;
  v8f z = {0.f,0.f,0.f,0.f,0.f,0.f,0.f,0.f};
  v8f acc[4] = {z, z, z, z};
  const bf16_t* Arow  = A  + (size_t)(rt * 16 + idx) * Cc + kb;
  const bf16_t* Bbase = Wt + (size_t)(ct * 64 + idx) * Cc + kb;
  gemm4(Arow, Bbase, Bbase + 16 * Cc, Bbase + 32 * Cc, Bbase + 48 * Cc, Cc, acc);
#pragma unroll
  for (int d = 0; d < 4; d++) {
    int c = ct * 64 + d * 16 + idx;
    float bv = bias[c];
    int hh = c >> 6, dh = c & 63;
#pragma unroll
    for (int r = 0; r < 8; r++) {
      int R = rt * 16 + r + 8 * hv;
      int bb = R / NLt, t = R - bb * NLt;
      kk[(((size_t)(bb * Hh + hh) * NLt + t) * DHd + dh)] = (bf16_t)(acc[d][r] + bv);
    }
  }
}

// V = tokens @ Wv + bv -> v_t[b,h,dh,t] bf16 (token stride NLP)
__global__ __launch_bounds__(32)
void gemm_v_kernel(const bf16_t* __restrict__ A, const bf16_t* __restrict__ Wt,
                   const float* __restrict__ bias, bf16_t* __restrict__ vt) {
  int lane = threadIdx.x, hv = lane >> 4, idx = lane & 15, kb = hv * 8;
  int ct = blockIdx.x, rt = blockIdx.y;
  v8f z = {0.f,0.f,0.f,0.f,0.f,0.f,0.f,0.f};
  v8f acc[4] = {z, z, z, z};
  const bf16_t* Arow  = A  + (size_t)(rt * 16 + idx) * Cc + kb;
  const bf16_t* Bbase = Wt + (size_t)(ct * 64 + idx) * Cc + kb;
  gemm4(Arow, Bbase, Bbase + 16 * Cc, Bbase + 32 * Cc, Bbase + 48 * Cc, Cc, acc);
#pragma unroll
  for (int d = 0; d < 4; d++) {
    int c = ct * 64 + d * 16 + idx;
    float bv = bias[c];
    int hh = c >> 6, dh = c & 63;
#pragma unroll
    for (int r = 0; r < 8; r++) {
      int R = rt * 16 + r + 8 * hv;
      int bb = R / NLt, t = R - bb * NLt;
      vt[(((size_t)(bb * Hh + hh) * DHd + dh) * NLP + t)] = (bf16_t)(acc[d][r] + bv);
    }
  }
}

// x1 = xn + att @ Wp + bp   (f32)
__global__ __launch_bounds__(32)
void gemm_p_kernel(const bf16_t* __restrict__ A, const bf16_t* __restrict__ Wt,
                   const float* __restrict__ bias, const float* __restrict__ xn,
                   float* __restrict__ x1) {
  int lane = threadIdx.x, hv = lane >> 4, idx = lane & 15, kb = hv * 8;
  int ct = blockIdx.x, rt = blockIdx.y;
  v8f z = {0.f,0.f,0.f,0.f,0.f,0.f,0.f,0.f};
  v8f acc[4] = {z, z, z, z};
  const bf16_t* Arow  = A  + (size_t)(rt * 16 + idx) * Cc + kb;
  const bf16_t* Bbase = Wt + (size_t)(ct * 64 + idx) * Cc + kb;
  gemm4(Arow, Bbase, Bbase + 16 * Cc, Bbase + 32 * Cc, Bbase + 48 * Cc, Cc, acc);
#pragma unroll
  for (int d = 0; d < 4; d++) {
    int c = ct * 64 + d * 16 + idx;
    float bv = bias[c];
#pragma unroll
    for (int r = 0; r < 8; r++) {
      size_t R = (size_t)(rt * 16 + r + 8 * hv);
      x1[R * Cc + c] = xn[R * Cc + c] + acc[d][r] + bv;
    }
  }
}

// g = gelu(h @ W1 + b1)  (bf16, exact erf gelu)
__global__ __launch_bounds__(32)
void gemm_m1_kernel(const bf16_t* __restrict__ A, const bf16_t* __restrict__ Wt,
                    const float* __restrict__ bias, bf16_t* __restrict__ gout) {
  int lane = threadIdx.x, hv = lane >> 4, idx = lane & 15, kb = hv * 8;
  int ct = blockIdx.x, rt = blockIdx.y;
  v8f z = {0.f,0.f,0.f,0.f,0.f,0.f,0.f,0.f};
  v8f acc[4] = {z, z, z, z};
  const bf16_t* Arow  = A  + (size_t)(rt * 16 + idx) * Cc + kb;
  const bf16_t* Bbase = Wt + (size_t)(ct * 64 + idx) * Cc + kb;
  gemm4(Arow, Bbase, Bbase + 16 * Cc, Bbase + 32 * Cc, Bbase + 48 * Cc, Cc, acc);
#pragma unroll
  for (int d = 0; d < 4; d++) {
    int c = ct * 64 + d * 16 + idx;
    float bv = bias[c];
#pragma unroll
    for (int r = 0; r < 8; r++) {
      size_t R = (size_t)(rt * 16 + r + 8 * hv);
      float xg = acc[d][r] + bv;
      float y = 0.5f * xg * (1.0f + erff(xg * 0.70710678118654752f));
      gout[R * MLPd + c] = (bf16_t)y;
    }
  }
}

// out = x1 + g @ W2 + b2  (f32) ; K = 3072
__global__ __launch_bounds__(32)
void gemm_m2_kernel(const bf16_t* __restrict__ A, const bf16_t* __restrict__ Wt,
                    const float* __restrict__ bias, const float* __restrict__ x1,
                    float* __restrict__ out) {
  int lane = threadIdx.x, hv = lane >> 4, idx = lane & 15, kb = hv * 8;
  int ct = blockIdx.x, rt = blockIdx.y;
  v8f z = {0.f,0.f,0.f,0.f,0.f,0.f,0.f,0.f};
  v8f acc[4] = {z, z, z, z};
  const bf16_t* Arow  = A  + (size_t)(rt * 16 + idx) * MLPd + kb;
  const bf16_t* Bbase = Wt + (size_t)(ct * 64 + idx) * MLPd + kb;
  gemm4(Arow, Bbase, Bbase + 16 * MLPd, Bbase + 32 * MLPd, Bbase + 48 * MLPd,
        MLPd, acc);
#pragma unroll
  for (int d = 0; d < 4; d++) {
    int c = ct * 64 + d * 16 + idx;
    float bv = bias[c];
#pragma unroll
    for (int r = 0; r < 8; r++) {
      size_t R = (size_t)(rt * 16 + r + 8 * hv);
      out[R * Cc + c] = x1[R * Cc + c] + acc[d][r] + bv;
    }
  }
}

// ---------------------------------------------------------------------------
// Fused flash-style attention: one wave per (b, h, 16-row q tile).
// Streams 32-token tiles of K/V. On gfx1250 with async builtins, K/V tiles
// are double-buffered in LDS via global_load_async_to_lds_b128 (ASYNCcnt),
// prefetching tile t+1 while WMMAs consume tile t.
// ---------------------------------------------------------------------------
__global__ __launch_bounds__(32)
void attn_kernel(const bf16_t* __restrict__ q, const bf16_t* __restrict__ kk,
                 const bf16_t* __restrict__ vt, const int* __restrict__ mask,
                 bf16_t* __restrict__ att) {
  __shared__ bf16_t P[16 * 32];
#if HAS_ASYNC_LDS
  __shared__ bf16_t Kt[2][32 * 64];   // [buf][token][dh]   4KB each
  __shared__ bf16_t Vt[2][64 * 32];   // [buf][dh][token]   4KB each
#endif
  int lane = threadIdx.x, hv = lane >> 4, idx = lane & 15, kb = hv * 8;
  int rt = blockIdx.x;   // q row tile (N/16 = 64)
  int hh = blockIdx.y;   // head
  int bb = blockIdx.z;   // batch
  int n0 = rt * 16;

  const bf16_t* qbase = q  + ((size_t)(bb * Hh + hh) * Nn + n0) * DHd;
  const bf16_t* kbase = kk + (size_t)(bb * Hh + hh) * NLt * DHd;
  const bf16_t* vbase = vt + (size_t)(bb * Hh + hh) * DHd * NLP;
  const int*    mrow  = mask + ((size_t)bb * Nn + n0) * NLt;

#if HAS_ASYNC_LDS
  // Issue one 32-token K tile (32x64 bf16) + V tile (64x32 bf16) into buffer.
  auto issue_tile = [&](int t0, int buf) {
#pragma unroll
    for (int i = 0; i < 8; i++) {            // K: 256 16B chunks / 32 lanes
      int c = lane + 32 * i;
      int tok = c >> 3, part = c & 7;
      int t = t0 + tok; if (t > NLt - 1) t = NLt - 1;
      const bf16_t* gsrc = kbase + (size_t)t * DHd + part * 8;
      __builtin_amdgcn_global_load_async_to_lds_b128(
          (AS1 v4i*)gsrc, (AS3 v4i*)&Kt[buf][tok * 64 + part * 8], 0, 0);
    }
#pragma unroll
    for (int i = 0; i < 8; i++) {            // V: 256 16B chunks / 32 lanes
      int c = lane + 32 * i;
      int dh = c >> 2, part = c & 3;
      const bf16_t* gsrc = vbase + (size_t)dh * NLP + t0 + part * 8;
      __builtin_amdgcn_global_load_async_to_lds_b128(
          (AS1 v4i*)gsrc, (AS3 v4i*)&Vt[buf][dh * 32 + part * 8], 0, 0);
    }
  };
#endif

  // q A-fragments for K=0..31 and 32..63 (reused across all token tiles)
  const bf16_t* qrow = qbase + (size_t)idx * DHd + kb;
  v16bf qa0 = cat16(ld8(qrow),      ld8(qrow + 16));
  v16bf qa1 = cat16(ld8(qrow + 32), ld8(qrow + 48));

  float mrun[8], lrun[8];
  v8f o0 = {0.f,0.f,0.f,0.f,0.f,0.f,0.f,0.f};
  v8f o1 = o0, o2 = o0, o3 = o0;
#pragma unroll
  for (int r = 0; r < 8; r++) { mrun[r] = -1e30f; lrun[r] = 0.f; }

#if HAS_ASYNC_LDS
  issue_tile(0, 0);                          // prefetch first tile
#endif

  for (int t0 = 0; t0 < NLP; t0 += 32) {
#if HAS_ASYNC_LDS
    int bufc = (t0 >> 5) & 1;
    if (t0 + 32 < NLP) {
      issue_tile(t0 + 32, bufc ^ 1);         // prefetch next tile
      __builtin_amdgcn_s_wait_asynccnt(16);  // oldest 16 (current tile) done
    } else {
      __builtin_amdgcn_s_wait_asynccnt(0);
    }
    asm volatile("" ::: "memory");
    const bf16_t* kr0 = &Kt[bufc][(size_t)idx * 64 + kb];          // row stride 64
    const bf16_t* kr1 = &Kt[bufc][(size_t)(16 + idx) * 64 + kb];
#else
    int t0a = t0 + idx;      if (t0a > NLt - 1) t0a = NLt - 1;
    int t0b = t0 + 16 + idx; if (t0b > NLt - 1) t0b = NLt - 1;
    const bf16_t* kr0 = kbase + (size_t)t0a * DHd + kb;
    const bf16_t* kr1 = kbase + (size_t)t0b * DHd + kb;
#endif
    // ---- scores: two 16x16 column tiles, K=64 each (2 WMMAs/tile) ----
    v8f s0 = {0.f,0.f,0.f,0.f,0.f,0.f,0.f,0.f};
    v8f s1 = s0;
    {
      v16bf b0 = cat16(ld8(kr0),      ld8(kr0 + 16));
      v16bf b1 = cat16(ld8(kr0 + 32), ld8(kr0 + 48));
      s0 = wmma_bf16(qa0, b0, s0);
      s0 = wmma_bf16(qa1, b1, s0);
    }
    {
      v16bf b0 = cat16(ld8(kr1),      ld8(kr1 + 16));
      v16bf b1 = cat16(ld8(kr1 + 32), ld8(kr1 + 48));
      s1 = wmma_bf16(qa0, b0, s1);
      s1 = wmma_bf16(qa1, b1, s1);
    }
    // ---- mask + scale ----
    float sv0[8], sv1[8];
#pragma unroll
    for (int r = 0; r < 8; r++) {
      int R = r + 8 * hv;                          // row within tile
      int t = t0 + idx;
      int m0v = (t < NLt) ? mrow[(size_t)R * NLt + t] : 0;
      sv0[r] = m0v ? s0[r] * 0.125f : -1e30f;      // scale = DH^-0.5 = 1/8
      t = t0 + 16 + idx;
      int m1v = (t < NLt) ? mrow[(size_t)R * NLt + t] : 0;
      sv1[r] = m1v ? s1[r] * 0.125f : -1e30f;
    }
    // ---- online softmax (per row: reduce across 16 lanes of half-wave) ----
#pragma unroll
    for (int r = 0; r < 8; r++) {
      float mx = fmaxf(sv0[r], sv1[r]);
      for (int off = 1; off < 16; off <<= 1)
        mx = fmaxf(mx, __shfl_xor(mx, off, 32));
      float mnew = fmaxf(mrun[r], mx);
      float corr = __expf(mrun[r] - mnew);
      float p0 = __expf(sv0[r] - mnew);
      float p1 = __expf(sv1[r] - mnew);
      float ps = p0 + p1;
      for (int off = 1; off < 16; off <<= 1)
        ps += __shfl_xor(ps, off, 32);
      lrun[r] = lrun[r] * corr + ps;
      mrun[r] = mnew;
      o0[r] *= corr; o1[r] *= corr; o2[r] *= corr; o3[r] *= corr;
      int R = r + 8 * hv;
      P[R * 32 + idx]      = (bf16_t)p0;
      P[R * 32 + 16 + idx] = (bf16_t)p1;
    }
    __syncthreads();
    // ---- P (16x32) @ V (32x64): 4 WMMAs over dh tiles ----
    const bf16_t* pr = &P[idx * 32 + kb];
    v16bf pa = cat16(ld8(pr), ld8(pr + 16));
#pragma unroll
    for (int d = 0; d < 4; d++) {
#if HAS_ASYNC_LDS
      const bf16_t* vr = &Vt[bufc][(size_t)(d * 16 + idx) * 32 + kb];
#else
      const bf16_t* vr = vbase + (size_t)(d * 16 + idx) * NLP + t0 + kb;
#endif
      v16bf vb = cat16(ld8(vr), ld8(vr + 16));
      if      (d == 0) o0 = wmma_bf16(pa, vb, o0);
      else if (d == 1) o1 = wmma_bf16(pa, vb, o1);
      else if (d == 2) o2 = wmma_bf16(pa, vb, o2);
      else             o3 = wmma_bf16(pa, vb, o3);
    }
    __syncthreads();
  }
  // ---- normalize + store att[b, n, h*64+dh] bf16 ----
#pragma unroll
  for (int r = 0; r < 8; r++) {
    float inv = 1.0f / lrun[r];
    size_t R = (size_t)(n0 + r + 8 * hv);
    size_t base = ((size_t)bb * Nn + R) * Cc + (size_t)hh * DHd + idx;
    att[base + 0]  = (bf16_t)(o0[r] * inv);
    att[base + 16] = (bf16_t)(o1[r] * inv);
    att[base + 32] = (bf16_t)(o2[r] * inv);
    att[base + 48] = (bf16_t)(o3[r] * inv);
  }
}

// ---------------------------------------------------------------------------
// Host orchestration
// ---------------------------------------------------------------------------
extern "C" void kernel_launch(void* const* d_in, const int* in_sizes, int n_in,
                              void* d_out, int out_size, void* d_ws, size_t ws_size,
                              hipStream_t stream) {
  const float* x    = (const float*)d_in[0];
  const float* ce   = (const float*)d_in[1];
  const int*   mask = (const int*)  d_in[2];
  const float* ln1g = (const float*)d_in[3];
  const float* ln1b = (const float*)d_in[4];
  const float* ln2g = (const float*)d_in[5];
  const float* ln2b = (const float*)d_in[6];
  const float* Wq   = (const float*)d_in[7];
  const float* bq   = (const float*)d_in[8];
  const float* Wk   = (const float*)d_in[9];
  const float* bk   = (const float*)d_in[10];
  const float* Wv   = (const float*)d_in[11];
  const float* bv   = (const float*)d_in[12];
  const float* Wp   = (const float*)d_in[13];
  const float* bp   = (const float*)d_in[14];
  const float* W1   = (const float*)d_in[15];
  const float* b1   = (const float*)d_in[16];
  const float* W2   = (const float*)d_in[17];
  const float* b2   = (const float*)d_in[18];
  float* out = (float*)d_out;

  char* ws = (char*)d_ws;
  float*  xn     = (float*)(ws + 0);            // 50331648 B
  float*  x1     = (float*)(ws + 50331648);     // 50331648 B
  bf16_t* xnb    = (bf16_t*)(ws + 100663296);   // 25165824 B
  bf16_t* att    = (bf16_t*)(ws + 125829120);   // 25165824 B
  bf16_t* hbuf   = (bf16_t*)(ws + 150994944);   // 25165824 B
  bf16_t* tokens = (bf16_t*)(ws + 176160768);   // 32864256 B
  bf16_t* qb     = (bf16_t*)(ws + 209025024);   // 25165824 B
  bf16_t* kbuf   = (bf16_t*)(ws + 234190848);   // 32858112 B
  bf16_t* vtb    = (bf16_t*)(ws + 267048960);   // 33030144 B
  bf16_t* gbuf   = (bf16_t*)(ws + 176160768);   // aliases tokens/q/k/vt (dead)
  bf16_t* wqt    = (bf16_t*)(ws + 300079104);
  bf16_t* wkt    = (bf16_t*)(ws + 301258752);
  bf16_t* wvt    = (bf16_t*)(ws + 302438400);
  bf16_t* wpt    = (bf16_t*)(ws + 303618048);
  bf16_t* w1t    = (bf16_t*)(ws + 304797696);   // [3072][768]
  bf16_t* w2t    = (bf16_t*)(ws + 309516288);   // [768][3072]

  ln1_kernel<<<Bb * Nn, 256, 0, stream>>>(x, ln1g, ln1b, xn, xnb, tokens);
  color_kernel<<<(Bb * Ll * Cc + 255) / 256, 256, 0, stream>>>(ce, tokens);

  convw_kernel<<<(Cc * Cc + 255) / 256, 256, 0, stream>>>(Wq, wqt, Cc, Cc);
  convw_kernel<<<(Cc * Cc + 255) / 256, 256, 0, stream>>>(Wk, wkt, Cc, Cc);
  convw_kernel<<<(Cc * Cc + 255) / 256, 256, 0, stream>>>(Wv, wvt, Cc, Cc);
  convw_kernel<<<(Cc * Cc + 255) / 256, 256, 0, stream>>>(Wp, wpt, Cc, Cc);
  convw_kernel<<<(Cc * MLPd + 255) / 256, 256, 0, stream>>>(W1, w1t, Cc, MLPd);
  convw_kernel<<<(MLPd * Cc + 255) / 256, 256, 0, stream>>>(W2, w2t, MLPd, Cc);

  gemm_q_kernel<<<dim3(Cc / 64, Bb * Nn / 16), 32, 0, stream>>>(xnb, wqt, bq, qb);
  gemm_k_kernel<<<dim3(Cc / 64, Bb * NLt / 16), 32, 0, stream>>>(tokens, wkt, bk, kbuf);
  gemm_v_kernel<<<dim3(Cc / 64, Bb * NLt / 16), 32, 0, stream>>>(tokens, wvt, bv, vtb);

  attn_kernel<<<dim3(Nn / 16, Hh, Bb), 32, 0, stream>>>(qb, kbuf, vtb, mask, att);

  gemm_p_kernel<<<dim3(Cc / 64, Bb * Nn / 16), 32, 0, stream>>>(att, wpt, bp, xn, x1);
  ln2_kernel<<<Bb * Nn, 256, 0, stream>>>(x1, ln2g, ln2b, hbuf);
  gemm_m1_kernel<<<dim3(MLPd / 64, Bb * Nn / 16), 32, 0, stream>>>(hbuf, w1t, b1, gbuf);
  gemm_m2_kernel<<<dim3(Cc / 64, Bb * Nn / 16), 32, 0, stream>>>(gbuf, w2t, b2, x1, out);
}
